// SNetRMRR_32358283608572
// MI455X (gfx1250) — matrix operand order repaired and checked by reading, not compiled, not run on hardware
//
#include <hip/hip_runtime.h>

// ---------------------------------------------------------------------------
// SpherNet rigid-registration head on MI455X (gfx1250), compile-only target.
//  * conv = fused one-ring gather + bf16 WMMA GEMM (v_wmma_f32_16x16x32_bf16,
//    f32 accumulate), templated on (CIN,COUT) so all gather addressing folds.
//  * activations keep an f32 master + bf16 shadow (written by bn_apply/pool),
//    so the conv gathers 8 contiguous bf16 per run => single b128 per run.
//  * weights are repacked once per layer into bf16 WMMA fragment order
//    Wp[kt][n][k%32] => each lane's B fragment is two aligned b128 loads.
//  * each wave computes NT (<=4) adjacent 16x16 N-tiles sharing one A frag.
// BN batch stats force a grid-wide reduction per layer -> 3 kernels per conv.
// Working set (~31MB incl. packed weights) stays resident in the 192MB L2.
// ---------------------------------------------------------------------------

typedef __attribute__((ext_vector_type(16))) __bf16 v16bf;
typedef __attribute__((ext_vector_type(8)))  __bf16 v8bf;
typedef __attribute__((ext_vector_type(8)))  float  v8f;

#define LRELU_SLOPE 0.2f
#define BN_EPS      1e-5f

// out[N x COUT] = gather7(hb, no) @ W + bias   (A-element k -> j=k/CIN, c=k%CIN)
template <int CIN, int COUT, int NT>
__global__ __launch_bounds__(32)
void snet_conv_wmma(const __bf16* __restrict__ hb,   // bf16 activations [N,CIN]
                    const int*    __restrict__ no,   // one-ring indices [N*7]
                    const __bf16* __restrict__ Wp,   // packed [K32][COUTP][32]
                    const float*  __restrict__ bias,
                    float* __restrict__ out, int N)
{
    constexpr int K      = 7 * CIN;
    constexpr int K32    = (K + 31) / 32;
    constexpr int COUTP  = (COUT + 15) & ~15;
    constexpr int TILESN = COUTP / 16;
    constexpr int GRIDN  = TILESN / NT;

    const int lane = threadIdx.x;              // wave32
    const int half = lane >> 4;
    const int l16  = lane & 15;
    const int tm   = blockIdx.x / GRIDN;
    const int tg   = blockIdx.x - tm * GRIDN;

    const int mA   = tm * 16 + l16;
    const int mrow = (mA < N) ? mA : 0;        // clamp for safe gather

    int  nB[NT];
    bool nOk[NT];
#pragma unroll
    for (int t = 0; t < NT; ++t) {
        nB[t]  = (tg * NT + t) * 16 + l16;
        nOk[t] = (COUT >= 16) ? true : (nB[t] < COUT);  // folds for COUT>=16
    }

    v8f acc[NT];
#pragma unroll
    for (int t = 0; t < NT; ++t) acc[t] = (v8f){};

    for (int kt = 0; kt < K32; ++kt) {
        const int k0 = kt * 32;

        // ---- A fragment, 16x32 bf16 (ISA 7.12.2):
        // lane half h needs k in [k0+8h, k0+8h+7] and [k0+16+8h, k0+16+8h+7]
        v16bf afrag;
        if constexpr (CIN >= 8) {
            // each 8-run starts at a multiple of 8 => single neighbor row,
            // 8 contiguous bf16 => one b128 load
            v8bf run0 = (v8bf){}, run1 = (v8bf){};
            {
                unsigned kb = (unsigned)(k0 + 8 * half);
                if (kb < (unsigned)K) {
                    unsigned j = kb / CIN, c = kb % CIN;
                    int nbj = no[mrow * 7 + j];
                    run0 = *(const v8bf*)(hb + (long)nbj * CIN + c);
                }
            }
            {
                unsigned kb = (unsigned)(k0 + 16 + 8 * half);
                if (kb < (unsigned)K) {
                    unsigned j = kb / CIN, c = kb % CIN;
                    int nbj = no[mrow * 7 + j];
                    run1 = *(const v8bf*)(hb + (long)nbj * CIN + c);
                }
            }
            afrag = __builtin_shufflevector(run0, run1,
                        0,1,2,3,4,5,6,7,8,9,10,11,12,13,14,15);
        } else {  // CIN==2 (block0 conv1 only): K=14, one chunk
#pragma unroll
            for (int e = 0; e < 16; ++e) {
                unsigned k = (unsigned)(k0 + 8 * half + e + ((e >= 8) ? 8 : 0));
                __bf16 v = (__bf16)0.0f;
                if (k < (unsigned)K) {
                    unsigned j = k / CIN, c = k % CIN;
                    v = hb[(long)no[mrow * 7 + j] * CIN + c];
                }
                afrag[e] = v;
            }
        }

        // ---- B fragments from packed weights: lane needs slots 8h..8h+7 and 16+8h..16+8h+7
        const __bf16* wbase = Wp + ((long)kt * COUTP) * 32 + 8 * half;
        if (kt + 1 < K32)  // speculative prefetch of next chunk
            __builtin_prefetch(wbase + (long)COUTP * 32, 0, 1);
#pragma unroll
        for (int t = 0; t < NT; ++t) {
            const v8bf* q = (const v8bf*)(wbase + (long)nB[t] * 32);
            v8bf lo = q[0], hi = q[2];
            v16bf bfrag = __builtin_shufflevector(lo, hi,
                              0,1,2,3,4,5,6,7,8,9,10,11,12,13,14,15);
            acc[t] = __builtin_amdgcn_wmma_f32_16x16x32_bf16(
                false, afrag, false, bfrag, (short)0, acc[t], false, false);
        }
    }

    // D layout: VGPR r -> M = r + 8*half ; N = lane&15
#pragma unroll
    for (int t = 0; t < NT; ++t) {
#pragma unroll
        for (int r = 0; r < 8; ++r) {
            int m = tm * 16 + r + 8 * half;
            if (m < N && nOk[t])
                out[(long)m * COUT + nB[t]] = acc[t][r] + bias[nB[t]];
        }
    }
}

// repack f32 weights [K,COUT] -> bf16 Wp[kt][n][k%32], zero padded (K, COUTP)
__global__ __launch_bounds__(256)
void snet_pack_w(const float* __restrict__ W, __bf16* __restrict__ Wp,
                 int K, int COUT, int COUTP, long total)
{
    long i = (long)blockIdx.x * 256 + threadIdx.x;
    if (i < total) {
        int  s  = (int)(i & 31);
        long r  = i >> 5;
        int  n  = (int)(r % COUTP);
        int  kt = (int)(r / COUTP);
        int  k  = kt * 32 + s;
        float v = (k < K && n < COUT) ? W[(long)k * COUT + n] : 0.0f;
        Wp[i] = (__bf16)v;
    }
}

// f32 -> bf16 shadow copy (for the network input x)
__global__ __launch_bounds__(256)
void snet_f2b(const float* __restrict__ src, __bf16* __restrict__ dst, long total)
{
    long i = (long)blockIdx.x * 256 + threadIdx.x;
    if (i < total) dst[i] = (__bf16)src[i];
}

// One workgroup per channel: biased batch stats -> scale/shift for BN.
__global__ __launch_bounds__(256)
void snet_bn_stats(const float* __restrict__ x, int N, int C,
                   const float* __restrict__ g, const float* __restrict__ be,
                   float* __restrict__ scale, float* __restrict__ shift)
{
    const int c = blockIdx.x;
    float s = 0.0f, s2 = 0.0f;
    for (int i = threadIdx.x; i < N; i += 256) {
        float v = x[(long)i * C + c];
        s += v; s2 += v * v;
    }
    __shared__ float sh[256], sh2[256];
    sh[threadIdx.x] = s; sh2[threadIdx.x] = s2;
    __syncthreads();
    for (int o = 128; o > 0; o >>= 1) {
        if (threadIdx.x < o) {
            sh[threadIdx.x]  += sh[threadIdx.x + o];
            sh2[threadIdx.x] += sh2[threadIdx.x + o];
        }
        __syncthreads();
    }
    if (threadIdx.x == 0) {
        float mean = sh[0] / (float)N;
        float var  = sh2[0] / (float)N - mean * mean;
        float sc   = g[c] * rsqrtf(var + BN_EPS);
        scale[c] = sc;
        shift[c] = be[c] - mean * sc;
    }
}

// y = lrelu(x*scale[c] + shift[c]) in place, plus bf16 shadow for next conv
__global__ __launch_bounds__(256)
void snet_bn_apply(float* __restrict__ x, __bf16* __restrict__ xb,
                   long total, int C,
                   const float* __restrict__ scale, const float* __restrict__ shift)
{
    long i = (long)blockIdx.x * 256 + threadIdx.x;
    if (i < total) {
        int   c = (int)(i % C);
        float v = x[i] * scale[c] + shift[c];
        v = (v >= 0.0f) ? v : LRELU_SLOPE * v;
        x[i]  = v;
        xb[i] = (__bf16)v;
    }
}

// mean-pool fine->coarse over the fine level 7-neighborhood (+ bf16 shadow)
__global__ __launch_bounds__(256)
void snet_pool(const float* __restrict__ hf, const int* __restrict__ no,
               float* __restrict__ hc, __bf16* __restrict__ hcb, int nC, int C)
{
    long i = (long)blockIdx.x * 256 + threadIdx.x;
    if (i < (long)nC * C) {
        int v = (int)(i / C), c = (int)(i % C);
        float s = 0.0f;
#pragma unroll
        for (int j = 0; j < 7; ++j) s += hf[(long)no[v * 7 + j] * C + c];
        float m = s * (1.0f / 7.0f);
        hc[i]  = m;
        hcb[i] = (__bf16)m;
    }
}

// channel-mean of [12,1024] -> 3 Euler angles -> rotation matrix R
__global__ __launch_bounds__(256)
void snet_head(const float* __restrict__ h, const float* __restrict__ oW,
               const float* __restrict__ ob, float* __restrict__ Rout,
               float* __restrict__ Rws)
{
    __shared__ float red[256];
    __shared__ float r[12];
    for (int row = 0; row < 12; ++row) {
        float s = 0.0f;
        for (int c = threadIdx.x; c < 1024; c += 256) s += h[row * 1024 + c];
        red[threadIdx.x] = s;
        __syncthreads();
        for (int o = 128; o > 0; o >>= 1) {
            if (threadIdx.x < o) red[threadIdx.x] += red[threadIdx.x + o];
            __syncthreads();
        }
        if (threadIdx.x == 0) r[row] = red[0] * (1.0f / 1024.0f);
        __syncthreads();
    }
    if (threadIdx.x == 0) {
        float ang[3];
        for (int j = 0; j < 3; ++j) {
            float s = ob[j];
            for (int k = 0; k < 12; ++k) s += r[k] * oW[j * 12 + k];
            ang[j] = s;
        }
        float cz = cosf(ang[0]), sz = sinf(ang[0]);
        float cy = cosf(ang[1]), sy = sinf(ang[1]);
        float cx = cosf(ang[2]), sx = sinf(ang[2]);
        float R[9] = {
            cz * cy, cz * sy * sx - sz * cx, sz * sx + cz * cx * sy,
            cy * sz, cz * cx + sz * sy * sx, cx * sz * sy - cz * sx,
            -sy,     cy * sx,                cy * cx
        };
        for (int t = 0; t < 9; ++t) { Rout[t] = R[t]; Rws[t] = R[t]; }
    }
}

// moved = xyz @ R.T
__global__ __launch_bounds__(256)
void snet_rot(const float* __restrict__ xyz, const float* __restrict__ R,
              float* __restrict__ out, int N)
{
    int i = blockIdx.x * 256 + threadIdx.x;
    if (i < N) {
        float x = xyz[3 * i], y = xyz[3 * i + 1], z = xyz[3 * i + 2];
        out[3 * i + 0] = R[0] * x + R[1] * y + R[2] * z;
        out[3 * i + 1] = R[3] * x + R[4] * y + R[5] * z;
        out[3 * i + 2] = R[6] * x + R[7] * y + R[8] * z;
    }
}

template <int CIN, int COUT>
static inline void conv_layer(const __bf16* hb, const int* no, const float* W,
                              const float* b, float* out, __bf16* Wp, int n,
                              hipStream_t s)
{
    constexpr int K      = 7 * CIN;
    constexpr int K32    = (K + 31) / 32;
    constexpr int COUTP  = (COUT + 15) & ~15;
    constexpr int TILESN = COUTP / 16;
    constexpr int NT     = (TILESN >= 4) ? 4 : TILESN;
    const long tot = (long)K32 * COUTP * 32;
    snet_pack_w<<<(int)((tot + 255) / 256), 256, 0, s>>>(W, Wp, K, COUT, COUTP, tot);
    const int grid = ((n + 15) / 16) * (TILESN / NT);
    snet_conv_wmma<CIN, COUT, NT><<<grid, 32, 0, s>>>(hb, no, Wp, b, out, n);
}

extern "C" void kernel_launch(void* const* d_in, const int* in_sizes, int n_in,
                              void* d_out, int out_size, void* d_ws, size_t ws_size,
                              hipStream_t stream)
{
    (void)n_in; (void)out_size; (void)ws_size;
    static const int NVv[8] = {163842, 40962, 10242, 2562, 642, 162, 42, 12};
    static const int CH[9]  = {2, 8, 16, 32, 64, 128, 256, 512, 1024};

    const float* x   = (const float*)d_in[0];
    const float* xyz = (const float*)d_in[1];
    const int*   no[8];
    for (int i = 0; i < 8; ++i) no[i] = (const int*)d_in[2 + i];
    const float* outrW = (const float*)d_in[74];
    const float* outrB = (const float*)d_in[75];

    // ---- workspace carve-up (256B aligned slabs), ~31MB total
    const long ELMAX = 163842L * 8;      // max activation elements
    char* cur = (char*)d_ws;
    auto take = [&cur](size_t bytes) {
        char* r = cur; cur += (bytes + 255) & ~(size_t)255; return r;
    };
    float*  bufXf = (float*) take(ELMAX * 4);
    float*  bufYf = (float*) take(ELMAX * 4);
    __bf16* bufXb = (__bf16*)take(ELMAX * 2);
    __bf16* bufYb = (__bf16*)take(ELMAX * 2);
    __bf16* xbf   = (__bf16*)take(163842L * 2 * 2);
    __bf16* Wp    = (__bf16*)take(224L * 1024 * 32 * 2);   // largest packed W
    float*  scale = (float*) take(1024 * 4);
    float*  shift = (float*) take(1024 * 4);
    float*  Rws   = (float*) take(64);

    // bf16 shadow of the input features
    snet_f2b<<<(int)((163842L * 2 + 255) / 256), 256, 0, stream>>>(x, xbf, 163842L * 2);

    const float*  hf = x;
    const __bf16* hb = xbf;
    for (int i = 0; i < 8; ++i) {
        const int n = NVv[i], cin = CH[i], cout = CH[i + 1];

        // locate block params; disambiguate dict flatten order by in_sizes
        const int base = 10 + 8 * i;
        const float *W1, *b1, *g1, *be1, *W2, *b2, *g2, *be2;
        if (in_sizes[base + 1] == cout) {
            W1 = (const float*)d_in[base + 0]; b1  = (const float*)d_in[base + 1];
            g1 = (const float*)d_in[base + 2]; be1 = (const float*)d_in[base + 3];
            W2 = (const float*)d_in[base + 4]; b2  = (const float*)d_in[base + 5];
            g2 = (const float*)d_in[base + 6]; be2 = (const float*)d_in[base + 7];
        } else {
            W1 = (const float*)d_in[base + 0]; W2  = (const float*)d_in[base + 1];
            b1 = (const float*)d_in[base + 2]; b2  = (const float*)d_in[base + 3];
            be1= (const float*)d_in[base + 4]; be2 = (const float*)d_in[base + 5];
            g1 = (const float*)d_in[base + 6]; g2  = (const float*)d_in[base + 7];
        }

        if (i > 0) {  // pool finer level into the free buffer (f32 + bf16)
            float*  dF = (hf == bufXf) ? bufYf : bufXf;
            __bf16* dB = (hf == bufXf) ? bufYb : bufXb;
            long tot = (long)n * cin;
            snet_pool<<<(int)((tot + 255) / 256), 256, 0, stream>>>(
                hf, no[i - 1], dF, dB, n, cin);
            hf = dF; hb = dB;
        }

        for (int half = 0; half < 2; ++half) {  // conv1 then conv2
            float*  dF = (hf == bufXf) ? bufYf : bufXf;
            __bf16* dB = (hf == bufXf) ? bufYb : bufXb;
            const float *W  = half ? W2  : W1,  *bb = half ? b2  : b1;
            const float *gg = half ? g2  : g1,  *bе = half ? be2 : be1;
            if (!half) {
                switch (i) {
                case 0: conv_layer<2,    8   >(hb, no[i], W, bb, dF, Wp, n, stream); break;
                case 1: conv_layer<8,    16  >(hb, no[i], W, bb, dF, Wp, n, stream); break;
                case 2: conv_layer<16,   32  >(hb, no[i], W, bb, dF, Wp, n, stream); break;
                case 3: conv_layer<32,   64  >(hb, no[i], W, bb, dF, Wp, n, stream); break;
                case 4: conv_layer<64,   128 >(hb, no[i], W, bb, dF, Wp, n, stream); break;
                case 5: conv_layer<128,  256 >(hb, no[i], W, bb, dF, Wp, n, stream); break;
                case 6: conv_layer<256,  512 >(hb, no[i], W, bb, dF, Wp, n, stream); break;
                case 7: conv_layer<512,  1024>(hb, no[i], W, bb, dF, Wp, n, stream); break;
                }
            } else {
                switch (i) {
                case 0: conv_layer<8,    8   >(hb, no[i], W, bb, dF, Wp, n, stream); break;
                case 1: conv_layer<16,   16  >(hb, no[i], W, bb, dF, Wp, n, stream); break;
                case 2: conv_layer<32,   32  >(hb, no[i], W, bb, dF, Wp, n, stream); break;
                case 3: conv_layer<64,   64  >(hb, no[i], W, bb, dF, Wp, n, stream); break;
                case 4: conv_layer<128,  128 >(hb, no[i], W, bb, dF, Wp, n, stream); break;
                case 5: conv_layer<256,  256 >(hb, no[i], W, bb, dF, Wp, n, stream); break;
                case 6: conv_layer<512,  512 >(hb, no[i], W, bb, dF, Wp, n, stream); break;
                case 7: conv_layer<1024, 1024>(hb, no[i], W, bb, dF, Wp, n, stream); break;
                }
            }
            snet_bn_stats<<<cout, 256, 0, stream>>>(dF, n, cout, gg, bе, scale, shift);
            long tot = (long)n * cout;
            snet_bn_apply<<<(int)((tot + 255) / 256), 256, 0, stream>>>(
                dF, dB, tot, cout, scale, shift);
            hf = dF; hb = dB;
        }
    }

    // head: hf is [12,1024] -> R (written to both d_out tail and workspace)
    float* moved = (float*)d_out;
    float* Rout  = moved + 163842L * 3;
    snet_head<<<1, 256, 0, stream>>>(hf, outrW, outrB, Rout, Rws);
    snet_rot<<<(163842 + 255) / 256, 256, 0, stream>>>(xyz, Rws, moved, 163842);
}
